// MHA_53841710023060
// MI455X (gfx1250) — matrix operand order repaired and checked by reading, not compiled, not run on hardware
//
#include <hip/hip_runtime.h>
#include <hip/hip_bf16.h>

#define BATCH 2
#define SEQ 2048
#define DMODEL 512
#define KDIM 256
#define VDIM 256
#define ODIM 512
#define NHEAD 8
#define HDK 32   // per-head key dim
#define TOK (BATCH*SEQ)   // 4096

typedef __attribute__((ext_vector_type(16))) __bf16 v16bf;
typedef __attribute__((ext_vector_type(8)))  __bf16 v8bf;
typedef __attribute__((ext_vector_type(8)))  float  v8f;

static __device__ __forceinline__ v16bf cat8(v8bf lo, v8bf hi) {
    return __builtin_shufflevector(lo, hi, 0,1,2,3,4,5,6,7,8,9,10,11,12,13,14,15);
}

static __device__ __forceinline__ v8f wmma_bf16(v16bf a, v16bf b, v8f c) {
    // (neg_a, A, neg_b, B, c_mod, C, reuse_a, reuse_b)
    return __builtin_amdgcn_wmma_f32_16x16x32_bf16(false, a, false, b, (short)0, c, false, false);
}

// ---------------- elementwise converts ----------------

__global__ void cvt_bf16_kernel(const float* __restrict__ in, __bf16* __restrict__ out, int n) {
    int i = blockIdx.x * blockDim.x + threadIdx.x;
    if (i < n) out[i] = (__bf16)in[i];
}

// in: [K, N] f32 row-major  ->  out: [N, K] bf16 row-major (i.e. W^T)
__global__ void cvt_t_bf16_kernel(const float* __restrict__ in, __bf16* __restrict__ out, int K, int N) {
    int i = blockIdx.x * blockDim.x + threadIdx.x;
    if (i < K * N) {
        int n = i / K;
        int k = i - n * K;
        out[i] = (__bf16)in[(size_t)k * N + n];
    }
}

// ---------------- generic WMMA GEMM ----------------
// C[M,N] = A[M,K] * W[K,N] + bias, with W given transposed (Wt[N,K] bf16).
// One wave (32 threads) computes a 16x64 output tile.
// mode 0: out = float [M,N], plain row-major
// mode 1: out = bf16 head-split [B, H, S, 32]  (M token-major, N = h*32+j)
// mode 2: out = bf16 v-transposed [B, H, 32, S]
__global__ __launch_bounds__(32)
void gemm_wmma_kernel(const __bf16* __restrict__ A, const __bf16* __restrict__ Wt,
                      const float* __restrict__ bias, void* __restrict__ out,
                      int M, int N, int K, int mode) {
    const int lane = threadIdx.x;          // 0..31
    const int row  = lane & 15;
    const int hi   = lane >> 4;            // 0/1: lane half
    const int mt   = blockIdx.x;           // M/16 tiles
    const int nt   = blockIdx.y;           // N/64 tiles

    v8f c[4] = {v8f{}, v8f{}, v8f{}, v8f{}};

    const __bf16* arow = A + (size_t)(mt * 16 + row) * K;
    for (int kk = 0; kk < K; kk += 32) {
        // A fragment: contiguous K -> lanes0-15: K 0..15; lanes16-31: K 16..31
        v16bf af = *(const v16bf*)(arow + kk + hi * 16);
        #pragma unroll
        for (int j = 0; j < 4; ++j) {
            // B fragment from Wt[N,K]: lane = column (n), contiguous 16 K-values per lane-half
            const __bf16* wrow = Wt + (size_t)(nt * 64 + j * 16 + row) * K + kk + hi * 16;
            v16bf bf = *(const v16bf*)wrow;
            c[j] = wmma_bf16(af, bf, c[j]);
        }
    }

    // C layout: VGPR g holds row (g + 8*hi), column = lane&15 (per 16x16 f32 table)
    #pragma unroll
    for (int j = 0; j < 4; ++j) {
        #pragma unroll
        for (int g = 0; g < 8; ++g) {
            int m = mt * 16 + g + hi * 8;
            int n = nt * 64 + j * 16 + row;
            float v = c[j][g] + bias[n];
            if (mode == 0) {
                ((float*)out)[(size_t)m * N + n] = v;
            } else {
                int b = m / SEQ, s = m - (m / SEQ) * SEQ;
                int h = n >> 5, jj = n & 31;
                if (mode == 1)
                    ((__bf16*)out)[(((size_t)(b * NHEAD + h)) * SEQ + s) * HDK + jj] = (__bf16)v;
                else
                    ((__bf16*)out)[(((size_t)(b * NHEAD + h)) * HDK + jj) * SEQ + s] = (__bf16)v;
            }
        }
    }
}

// ---------------- flash attention ----------------
// qh, kh: [B, H, S, 32] bf16 ; vt: [B, H, 32, S] bf16 ; mask: [B,1,S,S] f32
// ctx out: [B, S, 256] bf16 (merged heads). One wave per (b, h, 16-query tile).
__global__ __launch_bounds__(32)
void attn_kernel(const __bf16* __restrict__ qh, const __bf16* __restrict__ kh,
                 const __bf16* __restrict__ vt, const float* __restrict__ mask,
                 __bf16* __restrict__ ctx) {
    __shared__ __bf16 lp[16 * 32];       // P tile staging: row-major [16 q][32 k]

    const int lane = threadIdx.x;
    const int row  = lane & 15;
    const int hi   = lane >> 4;
    const int qt = blockIdx.x;           // S/16
    const int h  = blockIdx.y;
    const int b  = blockIdx.z;
    const size_t bh = (size_t)b * NHEAD + h;

    // q A-fragment, loaded once and reused for all key tiles
    const __bf16* qrow = qh + (bh * SEQ + (size_t)qt * 16 + row) * HDK;
    v16bf qa = *(const v16bf*)(qrow + hi * 16);

    float mrow[8], lrow[8];
    #pragma unroll
    for (int g = 0; g < 8; ++g) { mrow[g] = -INFINITY; lrow[g] = 0.f; }
    v8f o0 = v8f{}, o1 = v8f{};

    const float rden = 0.0625f;          // 1/sqrt(KEY_DIM) = 1/16
    const float* mbase = mask + (size_t)b * SEQ * SEQ + (size_t)(qt * 16) * SEQ;

    for (int kb = 0; kb < SEQ; kb += 32) {
        // K B-fragments (kh row-major [key][32] == col-major of k^T)
        v16bf k0 = *(const v16bf*)(kh + (bh * SEQ + kb + row) * HDK + hi * 16);
        v16bf k1 = *(const v16bf*)(kh + (bh * SEQ + kb + 16 + row) * HDK + hi * 16);
        v8f s0 = wmma_bf16(qa, k0, v8f{});
        v8f s1 = wmma_bf16(qa, k1, v8f{});

        #pragma unroll
        for (int g = 0; g < 8; ++g) {
            const float* mr = mbase + (size_t)(g + hi * 8) * SEQ + kb + row;
            float x0 = s0[g] * rden + mr[0];
            float x1 = s1[g] * rden + mr[16];
            // row max across the 16 lanes holding this row
            float t = fmaxf(x0, x1);
            t = fmaxf(t, __shfl_xor(t, 1));
            t = fmaxf(t, __shfl_xor(t, 2));
            t = fmaxf(t, __shfl_xor(t, 4));
            t = fmaxf(t, __shfl_xor(t, 8));
            float nm = fmaxf(mrow[g], t);
            float al = __expf(mrow[g] - nm);
            mrow[g] = nm;
            float p0 = __expf(x0 - nm);
            float p1 = __expf(x1 - nm);
            float r = p0 + p1;
            r += __shfl_xor(r, 1);
            r += __shfl_xor(r, 2);
            r += __shfl_xor(r, 4);
            r += __shfl_xor(r, 8);
            lrow[g] = lrow[g] * al + r;
            o0[g] *= al;
            o1[g] *= al;
            lp[(g + hi * 8) * 32 + row]      = (__bf16)p0;
            lp[(g + hi * 8) * 32 + row + 16] = (__bf16)p1;
        }
        __syncthreads();

        // reload P in A-fragment layout from LDS
        const __bf16* prow = &lp[row * 32];
        v16bf pa = *(const v16bf*)(prow + hi * 16);
        // V B-fragments from vt [B,H,32,S]: lane = dv column, contiguous keys
        v16bf v0 = *(const v16bf*)(vt + (bh * HDK + row) * SEQ + kb + hi * 16);
        v16bf v1 = *(const v16bf*)(vt + (bh * HDK + 16 + row) * SEQ + kb + hi * 16);
        o0 = wmma_bf16(pa, v0, o0);
        o1 = wmma_bf16(pa, v1, o1);
        __syncthreads();
    }

    #pragma unroll
    for (int g = 0; g < 8; ++g) {
        float inv = 1.f / lrow[g];
        int m = qt * 16 + g + hi * 8;
        size_t base = ((size_t)b * SEQ + m) * (NHEAD * HDK) + h * HDK;
        ctx[base + row]      = (__bf16)(o0[g] * inv);
        ctx[base + 16 + row] = (__bf16)(o1[g] * inv);
    }
}

// ---------------- host launch ----------------

extern "C" void kernel_launch(void* const* d_in, const int* in_sizes, int n_in,
                              void* d_out, int out_size, void* d_ws, size_t ws_size,
                              hipStream_t stream) {
    (void)in_sizes; (void)n_in; (void)out_size; (void)ws_size;
    const float* V    = (const float*)d_in[0];
    const float* Q    = (const float*)d_in[1];
    const float* K    = (const float*)d_in[2];
    const float* mask = (const float*)d_in[3];
    const float* Wq   = (const float*)d_in[4];
    const float* bq   = (const float*)d_in[5];
    const float* Wk   = (const float*)d_in[6];
    const float* bk   = (const float*)d_in[7];
    const float* Wv   = (const float*)d_in[8];
    const float* bv   = (const float*)d_in[9];
    const float* Wo   = (const float*)d_in[10];
    const float* bo   = (const float*)d_in[11];

    char* ws = (char*)d_ws;
    size_t off = 0;
    auto carve = [&](size_t bytes) { char* p = ws + off; off += (bytes + 255) & ~(size_t)255; return p; };

    __bf16* Qbf  = (__bf16*)carve((size_t)TOK * DMODEL * 2);   // 4 MB
    __bf16* Kbf  = (__bf16*)carve((size_t)TOK * DMODEL * 2);
    __bf16* Vbf  = (__bf16*)carve((size_t)TOK * DMODEL * 2);
    __bf16* Wqt  = (__bf16*)carve((size_t)KDIM * DMODEL * 2);
    __bf16* Wkt  = (__bf16*)carve((size_t)KDIM * DMODEL * 2);
    __bf16* Wvt  = (__bf16*)carve((size_t)VDIM * DMODEL * 2);
    __bf16* Wot  = (__bf16*)carve((size_t)ODIM * VDIM * 2);
    __bf16* qhd  = (__bf16*)carve((size_t)TOK * KDIM * 2);     // [B,H,S,32]
    __bf16* khd  = (__bf16*)carve((size_t)TOK * KDIM * 2);
    __bf16* vtd  = (__bf16*)carve((size_t)TOK * VDIM * 2);     // [B,H,32,S]
    __bf16* ctx  = (__bf16*)carve((size_t)TOK * VDIM * 2);     // [B,S,256]

    const int CT = 256;
    int nQKV = TOK * DMODEL;
    cvt_bf16_kernel<<<(nQKV + CT - 1) / CT, CT, 0, stream>>>(Q, Qbf, nQKV);
    cvt_bf16_kernel<<<(nQKV + CT - 1) / CT, CT, 0, stream>>>(K, Kbf, nQKV);
    cvt_bf16_kernel<<<(nQKV + CT - 1) / CT, CT, 0, stream>>>(V, Vbf, nQKV);

    int nW = DMODEL * KDIM;
    cvt_t_bf16_kernel<<<(nW + CT - 1) / CT, CT, 0, stream>>>(Wq, Wqt, DMODEL, KDIM);
    cvt_t_bf16_kernel<<<(nW + CT - 1) / CT, CT, 0, stream>>>(Wk, Wkt, DMODEL, KDIM);
    cvt_t_bf16_kernel<<<(nW + CT - 1) / CT, CT, 0, stream>>>(Wv, Wvt, DMODEL, VDIM);
    cvt_t_bf16_kernel<<<(nW + CT - 1) / CT, CT, 0, stream>>>(Wo, Wot, VDIM, ODIM);

    // projections: M=4096, N=256, K=512
    dim3 gp(TOK / 16, KDIM / 64);
    gemm_wmma_kernel<<<gp, 32, 0, stream>>>(Qbf, Wqt, bq, (void*)qhd, TOK, KDIM, DMODEL, 1);
    gemm_wmma_kernel<<<gp, 32, 0, stream>>>(Kbf, Wkt, bk, (void*)khd, TOK, KDIM, DMODEL, 1);
    gemm_wmma_kernel<<<gp, 32, 0, stream>>>(Vbf, Wvt, bv, (void*)vtd, TOK, VDIM, DMODEL, 2);

    // attention
    dim3 ga(SEQ / 16, NHEAD, BATCH);
    attn_kernel<<<ga, 32, 0, stream>>>(qhd, khd, vtd, mask, ctx);

    // output projection: M=4096, N=512, K=256, f32 out
    dim3 go(TOK / 16, ODIM / 64);
    gemm_wmma_kernel<<<go, 32, 0, stream>>>(ctx, Wot, bo, d_out, TOK, ODIM, VDIM, 0);
}
